// LinearSystem_54193897341078
// MI455X (gfx1250) — matrix operand order repaired and checked by reading, not compiled, not run on hardware
//
#include <hip/hip_runtime.h>
#include <hip/hip_bf16.h>
#include <stdint.h>

// ---------------------------------------------------------------------------
// CDNA5 / gfx1250: wave32, WMMA 16x16x32 bf16 -> f32 accumulate, TDM staging.
// ---------------------------------------------------------------------------
typedef __attribute__((ext_vector_type(16))) __bf16 v16bf;
typedef __attribute__((ext_vector_type(8)))  __bf16 v8bf;
typedef __attribute__((ext_vector_type(4)))  __bf16 v4bf;
typedef __attribute__((ext_vector_type(8)))  float  v8f;
typedef __attribute__((ext_vector_type(4)))  float  f32x4;
typedef __attribute__((ext_vector_type(4)))  unsigned int u32x4;
typedef __attribute__((ext_vector_type(8)))  unsigned int u32x8;

static constexpr int SD = 256;       // state/input/obs dim
static constexpr int BATCH = 64;
static constexpr int TSTEPS = 2048;

__device__ inline v16bf cat8(v8bf lo, v8bf hi) {
  return __builtin_shufflevector(lo, hi, 0, 1, 2, 3, 4, 5, 6, 7,
                                         8, 9, 10, 11, 12, 13, 14, 15);
}
__device__ inline v8bf cvt8(f32x4 a, f32x4 b) {
  v8bf r;
  r[0] = (__bf16)a.x; r[1] = (__bf16)a.y; r[2] = (__bf16)a.z; r[3] = (__bf16)a.w;
  r[4] = (__bf16)b.x; r[5] = (__bf16)b.y; r[6] = (__bf16)b.z; r[7] = (__bf16)b.w;
  return r;
}

// Load a full 16x32 bf16 B-fragment (K rows x 16 cols) straight from a
// row-major f32 matrix M[N][K] used as M^T: in this code's k order, element e of lane l is
// M[n0 + l%16][k0 + (l/16)*16 + e] -> 16 *contiguous* floats per lane.
__device__ inline v16bf load_b_frag_global_f32(const float* __restrict__ M,
                                               int ldk, int n0, int k0, int lane) {
  const int n = n0 + (lane & 15);
  const float* src = &M[(long)n * ldk + k0 + (lane >> 4) * 16];
  const f32x4* s4 = (const f32x4*)src;
  return cat8(cvt8(s4[0], s4[1]), cvt8(s4[2], s4[3]));
}

// ---------------------------------------------------------------------------
// TDM: stage a [16 x 256] f32 tile (row stride = T*256 floats) into LDS.
// D# per ISA 08 §8.3/8.4: group0 {count=1, lds_addr, global_addr, type=2},
// group1 {data_size=4B, tensor 256 x 16, tile 256 x 16, dim0_stride=T*256}.
// Issued by one wave; tracked with TENSORcnt.
// ---------------------------------------------------------------------------
__device__ inline void tdm_load_u_tile(float* lds_dst, const float* gsrc, int T) {
  const uint64_t ga  = (uint64_t)(uintptr_t)gsrc;
  const uint32_t lds = (uint32_t)(uintptr_t)lds_dst;   // low 32 bits = LDS offset
  const u32x4 g0 = { 1u,                                   // count=1, user mode
                     lds,                                  // lds_addr
                     (uint32_t)ga,                         // global_addr[31:0]
                     (uint32_t)(ga >> 32) | (2u << 30) };  // addr[56:32] | type=2
  u32x8 g1;
  g1[0] = (2u << 16);            // workgroup_mask=0, data_size=2 (4 bytes)
  g1[1] = (256u << 16);          // tensor_dim0 = 256 (low16 at [63:48])
  g1[2] = (16u << 16);           // tensor_dim0 hi=0 ; tensor_dim1 = 16
  g1[3] = (256u << 16);          // tensor_dim1 hi=0 ; tile_dim0 = 256
  g1[4] = 16u;                   // tile_dim1 = 16 ; tile_dim2 = 0
  g1[5] = (uint32_t)T * 256u;    // tensor_dim0_stride low32
  g1[6] = 0u;                    // stride hi16 ; dim1_stride low16
  g1[7] = 0u;
  asm volatile("tensor_load_to_lds %0, %1" :: "s"(g0), "s"(g1) : "memory");
}

// ---------------------------------------------------------------------------
// dual_gemm_bf16: out[r,n] = sum_k X1[r,k]*W1[n,k] + sum_k X2[r,k]*W2[n,k]
// 256 threads = 8 waves; 32-row x 256-col tile per block.
// B tile staged in LDS in fragment-major order -> frag loads are 2x b128.
// ---------------------------------------------------------------------------
__global__ __launch_bounds__(256)
void dual_gemm_bf16(const float* __restrict__ X1, const float* __restrict__ W1,
                    const float* __restrict__ X2, const float* __restrict__ W2,
                    float* __restrict__ out) {
  constexpr int KD = SD, ND = SD, MT = 32, KT = 32;
  __shared__ __align__(16) __bf16 a_lds[MT * KT];        // 2 KB, row-major [m][k]
  __shared__ __align__(16) __bf16 b_pack[16 * 32 * 16];  // 16 KB, [ntile][lane][e]

  const int tid  = threadIdx.x;
  const int lane = tid & 31;
  const int wave = tid >> 5;
  const long row0 = (long)blockIdx.x * MT;
  const int mtile  = wave & 1;
  const int ntbase = (wave >> 1) * 4;          // 4 consecutive N tiles per wave
  const int g = lane >> 4, g8 = g * 8, nr = lane & 15, mr = lane & 15;

  v8f c[4];
#pragma unroll
  for (int j = 0; j < 4; ++j)
#pragma unroll
    for (int r = 0; r < 8; ++r) c[j][r] = 0.0f;

#pragma unroll 1
  for (int p = 0; p < 2; ++p) {
    const float* __restrict__ X = p ? X2 : X1;
    const float* __restrict__ W = p ? W2 : W1;
#pragma unroll 1
    for (int kk = 0; kk < KD; kk += KT) {
      __syncthreads();
      { // Stage A tile (32 rows x 32 k): one float4 load + one b64 store/thread
        const int m = tid >> 3, kl = (tid & 7) * 4;
        const f32x4 x = *(const f32x4*)&X[(row0 + m) * KD + kk + kl];
        v4bf h; h[0] = (__bf16)x.x; h[1] = (__bf16)x.y;
               h[2] = (__bf16)x.z; h[3] = (__bf16)x.w;
        *(v4bf*)&a_lds[m * KT + kl] = h;
      }
      // Stage B tile fragment-major: slot s=(ntile,lane) holds its 16 halves
#pragma unroll
      for (int ss = 0; ss < 2; ++ss) {
        const int s  = tid * 2 + ss;
        const int nt = s >> 5, ls = s & 31;
        const int n  = nt * 16 + (ls & 15);
        const float* src = &W[(long)n * KD + kk + (ls >> 4) * 16];
        const f32x4* s4 = (const f32x4*)src;
        v8bf* dst = (v8bf*)&b_pack[s * 16];
        dst[0] = cvt8(s4[0], s4[1]);
        dst[1] = cvt8(s4[2], s4[3]);
      }
      if (kk + KT < KD)
        __builtin_prefetch(&X[(row0 + (tid >> 3)) * KD + kk + KT], 0, 1);
      __syncthreads();

      // A fragment: two contiguous b128 LDS loads per lane
      const v8bf alo = *(const v8bf*)&a_lds[(mtile * 16 + mr) * KT + 2 * g8];
      const v8bf ahi = *(const v8bf*)&a_lds[(mtile * 16 + mr) * KT + 2 * g8 + 8];
      const v16bf a = cat8(alo, ahi);
#pragma unroll
      for (int j = 0; j < 4; ++j) {
        const v8bf* bp = (const v8bf*)&b_pack[((ntbase + j) * 32 + lane) * 16];
        const v16bf b = cat8(bp[0], bp[1]);
        c[j] = __builtin_amdgcn_wmma_f32_16x16x32_bf16(
            false, a, false, b, (short)0, c[j], false, false);
      }
    }
  }

  // Epilogue: C layout -> row = r + 8*(lane/16), col = lane%16 within tile.
#pragma unroll
  for (int j = 0; j < 4; ++j) {
    const int col = (ntbase + j) * 16 + nr;
#pragma unroll
    for (int r = 0; r < 8; ++r) {
      const long row = row0 + mtile * 16 + r + 8 * g;
      out[row * ND + col] = c[j][r];
    }
  }
}

// ---------------------------------------------------------------------------
// lds_scan: s_t = s_{t-1} @ F^T + u_t, in place on states (holds u on entry).
// Grid = 4 blocks x 16 batch rows, 256 threads = 8 waves.
//  - F^T lives entirely in VGPRs: 16 fragments/wave, loaded once from global
//    (fragment lanes map to 16 contiguous f32 of row-major F -> float4 loads).
//  - u_{t+1} is TDM-prefetched into a double-buffered LDS tile each step.
//  - bf16 state double-buffered in LDS; A-frag loads are 2x b128.
// ---------------------------------------------------------------------------
__global__ __launch_bounds__(256)
void lds_scan(const float* __restrict__ state0, const float* __restrict__ F,
              float* __restrict__ states, int T) {
  constexpr int BR = 16;  // batch rows per workgroup
  __shared__ __align__(16) __bf16 s_buf[2][BR * SD];  // 16 KB
  __shared__ __align__(16) float  u_lds[2][BR * SD];  // 32 KB

  const int tid  = threadIdx.x;
  const int lane = tid & 31;
  const int wave = tid >> 5;
  const int b0   = blockIdx.x * BR;
  const int g = lane >> 4, g8 = g * 8, nr = lane & 15, mr = lane & 15;
  const int n0a = wave * 32;  // two 16-wide N tiles per wave

  // F^T fragments held in registers for the whole scan (time-invariant B).
  v16bf bfrag[2][8];
#pragma unroll
  for (int j = 0; j < 2; ++j)
#pragma unroll
    for (int kc = 0; kc < 8; ++kc)
      bfrag[j][kc] = load_b_frag_global_f32(F, SD, n0a + j * 16, kc * 32, lane);

  // state0 -> s_buf[0] : 16 floats/thread, wide LDS stores
  {
    const int m = tid >> 4, k = (tid & 15) * 16;
    const f32x4* s4 = (const f32x4*)&state0[(long)(b0 + m) * SD + k];
    v8bf* dst = (v8bf*)&s_buf[0][m * SD + k];
    dst[0] = cvt8(s4[0], s4[1]);
    dst[1] = cvt8(s4[2], s4[3]);
  }

  // Kick off TDM for u_0.
  if (wave == 0)
    tdm_load_u_tile(&u_lds[0][0], &states[(long)b0 * T * SD + 0 * SD], T);
  __syncthreads();

#pragma unroll 1
  for (int t = 0; t < T; ++t) {
    const __bf16* s_cur = s_buf[t & 1];
    __bf16*       s_nxt = s_buf[(t + 1) & 1];
    const float*  ub    = u_lds[t & 1];

    // Prefetch u_{t+1} while waiting only for u_t (in-order TENSORcnt).
    if (wave == 0) {
      if (t + 1 < T) {
        tdm_load_u_tile(&u_lds[(t + 1) & 1][0],
                        &states[((long)b0 * T + (t + 1)) * SD], T);
        __builtin_amdgcn_s_wait_tensorcnt(1);
      } else {
        __builtin_amdgcn_s_wait_tensorcnt(0);
      }
    }
    __syncthreads();  // u_t visible to all waves

    // C init from u_t (LDS).
    v8f c[2];
#pragma unroll
    for (int j = 0; j < 2; ++j) {
      const int col = n0a + j * 16 + nr;
#pragma unroll
      for (int r = 0; r < 8; ++r) c[j][r] = ub[(r + 8 * g) * SD + col];
    }

    // s_{t-1} @ F^T : 8 K-chunks, B already in registers.
#pragma unroll
    for (int kc = 0; kc < 8; ++kc) {
      const v8bf alo = *(const v8bf*)&s_cur[mr * SD + kc * 32 + 2 * g8];
      const v8bf ahi = *(const v8bf*)&s_cur[mr * SD + kc * 32 + 2 * g8 + 8];
      const v16bf a = cat8(alo, ahi);
#pragma unroll
      for (int j = 0; j < 2; ++j)
        c[j] = __builtin_amdgcn_wmma_f32_16x16x32_bf16(
            false, a, false, bfrag[j][kc], (short)0, c[j], false, false);
    }

    // Write states_t (f32 global) and stage bf16 state for t+1.
#pragma unroll
    for (int j = 0; j < 2; ++j) {
      const int col = n0a + j * 16 + nr;
#pragma unroll
      for (int r = 0; r < 8; ++r) {
        const int ml = r + 8 * g;
        const float v = c[j][r];
        states[((long)(b0 + ml) * T + t) * SD + col] = v;
        s_nxt[ml * SD + col] = (__bf16)v;
      }
    }
    __syncthreads();  // s_nxt published before next step reads it
  }
}

// ---------------------------------------------------------------------------
// Launch: u = inputs@B^T + W_noise@S_W^T  ->  scan in-place  ->
//         obs = states@H^T + V_noise@S_V^T
// ---------------------------------------------------------------------------
extern "C" void kernel_launch(void* const* d_in, const int* in_sizes, int n_in,
                              void* d_out, int out_size, void* d_ws, size_t ws_size,
                              hipStream_t stream) {
  (void)in_sizes; (void)n_in; (void)out_size; (void)d_ws; (void)ws_size;
  const float* inputs  = (const float*)d_in[0];
  const float* state0  = (const float*)d_in[1];
  const float* W_noise = (const float*)d_in[2];
  const float* V_noise = (const float*)d_in[3];
  const float* F       = (const float*)d_in[4];
  const float* B_mat   = (const float*)d_in[5];
  const float* H       = (const float*)d_in[6];
  const float* sqS_W   = (const float*)d_in[7];
  const float* sqS_V   = (const float*)d_in[8];

  float* out_states = (float*)d_out;
  float* out_obs    = out_states + (long)BATCH * TSTEPS * SD;

  const long R = (long)BATCH * TSTEPS;  // 131072 GEMM rows

  dual_gemm_bf16<<<(int)(R / 32), 256, 0, stream>>>(inputs, B_mat, W_noise, sqS_W,
                                                    out_states);
  lds_scan<<<BATCH / 16, 256, 0, stream>>>(state0, F, out_states, TSTEPS);
  dual_gemm_bf16<<<(int)(R / 32), 256, 0, stream>>>(out_states, H, V_noise, sqS_V,
                                                    out_obs);
}
